// MultiHeadAttention_71416716198046
// MI455X (gfx1250) — matrix operand order repaired and checked by reading, not compiled, not run on hardware
//
#include <hip/hip_runtime.h>
#include <hip/hip_bf16.h>
#include <math.h>

// ---------------------------------------------------------------------------
// MHA: convert -> qkv proj -> causal flash attention -> output proj
// B=2, T=2048, C=1024, H=16, D=64.  Matmuls on v_wmma_f32_16x16x32_bf16.
// Staging: TDM tensor_load_to_lds (A tiles) + global_load_async_to_lds_b128
// (B tiles), both CDNA5-only data paths.
// ---------------------------------------------------------------------------

#define TSEQ   2048
#define NEMBD  1024
#define NHEAD  16
#define HDIM   64
#define BT     (2 * TSEQ)        // 4096 rows of x

typedef __bf16 v16bf __attribute__((ext_vector_type(16)));
typedef __bf16 v8bf  __attribute__((ext_vector_type(8)));
typedef __bf16 v4bf  __attribute__((ext_vector_type(4)));
typedef float  v8f   __attribute__((ext_vector_type(8)));
typedef float  v4f   __attribute__((ext_vector_type(4)));
typedef unsigned int u32x4 __attribute__((ext_vector_type(4)));
typedef int    i32x4 __attribute__((ext_vector_type(4)));
typedef int    i32x8 __attribute__((ext_vector_type(8)));

#if __has_builtin(__builtin_amdgcn_tensor_load_to_lds)
#define HAS_TDM 1
#else
#define HAS_TDM 0
#endif

union FragBF {
    v16bf v;
    struct { v8bf lo, hi; } h;
};

// A-matrix fragment (16x32 bf16): lane = M row (mod 16); half-wave selects
// k-groups {klo..klo+7} (vgpr0-3) and {16+klo..16+klo+7} (vgpr4-7), klo=8*(lane/16).
__device__ inline v16bf load_frag_a(const __bf16* base, int stride, int lane) {
    int row = lane & 15;
    int klo = (lane >> 4) << 3;
    const __bf16* p = base + row * stride + klo;
    FragBF f;
    f.h.lo = *(const v8bf*)(p);
    f.h.hi = *(const v8bf*)(p + 16);
    return f.v;
}

// B-matrix fragment (32x16 bf16): lane = N col (mod 16); half-wave selects
// k-groups {klo..klo+7} and {16+klo..16+klo+7}, klo=8*(lane/16).  Source is n-major.
__device__ inline v16bf load_frag_b(const __bf16* base, int stride, int lane) {
    int n    = lane & 15;
    int koff = (lane >> 4) << 3;
    const __bf16* p = base + n * stride + koff;
    FragBF f;
    f.h.lo = *(const v8bf*)(p);
    f.h.hi = *(const v8bf*)(p + 16);
    return f.v;
}

__device__ inline v8f wmma_bf16(v16bf a, v16bf b, v8f c) {
    return __builtin_amdgcn_wmma_f32_16x16x32_bf16(false, a, false, b,
                                                   (short)0, c, false, false);
}

// CDNA5 async global->LDS copy (16B per lane), tracked by ASYNCcnt.
__device__ inline void async_b128(unsigned lds_off, const void* g) {
    asm volatile("global_load_async_to_lds_b128 %0, %1, off"
                 :: "v"(lds_off), "v"(g) : "memory");
}
__device__ inline void wait_async0() {
    asm volatile("s_wait_asynccnt 0x0" ::: "memory");
}
__device__ inline void wait_tensor0() {
#if __has_builtin(__builtin_amdgcn_s_wait_tensorcnt)
    __builtin_amdgcn_s_wait_tensorcnt((short)0);
#else
    asm volatile("s_wait_tensorcnt 0x0" ::: "memory");
#endif
}

// Stage `nchunks` row-chunks (8 bf16 = 16B each, 8 chunks per 64-elem row) from
// row-major global (leading dim ldg) into LDS rows padded to 72 elements.
__device__ inline void stage_rows_async(const __bf16* g, size_t ldg,
                                        unsigned lds_base, int nchunks, int tid) {
    for (int c = tid; c < nchunks; c += 256) {
        int row = c >> 3, seg = c & 7;
        async_b128(lds_base + (unsigned)(row * 72 + seg * 8) * 2,
                   g + (size_t)row * ldg + seg * 8);
    }
}

#if HAS_TDM
// TDM: DMA a 128x64 bf16 tile from a [4096][1024] bf16 tensor into LDS,
// padding 16B per 128B row so LDS row stride = 144B = 72 bf16 (frag layout).
__device__ inline void tdm_load_tile_a(const __bf16* gtile, unsigned lds_addr) {
    unsigned long long ga = (unsigned long long)(uintptr_t)gtile;
    u32x4 g0;
    g0[0] = 1u;                                   // count=1, user descriptor
    g0[1] = lds_addr;                             // lds_addr
    g0[2] = (unsigned)ga;                         // global_addr[31:0]
    g0[3] = ((unsigned)(ga >> 32) & 0x01FFFFFFu) | (2u << 30);  // type=2
    i32x8 g1;
    g1[0] = (1 << 16)            // data_size = 2 bytes
          | (1 << 20)            // pad_enable
          | (4 << 22)            // pad_interval: 32 dwords (=128B)
          | (3 << 25);           // pad_amount: 4 dwords (=16B)
    g1[1] = (int)(1024u << 16);  // tensor_dim0 = 1024 (low16 at bits 63:48)
    g1[2] = (int)(4096u << 16);  // dim0 hi16=0 | tensor_dim1 = 4096 (low16)
    g1[3] = (int)(64u << 16);    // dim1 hi16=0 | tile_dim0 = 64
    g1[4] = 128;                 // tile_dim1 = 128 | tile_dim2 = 0
    g1[5] = 1024;                // tensor_dim0_stride = 1024
    g1[6] = 0;                   // stride hi | tensor_dim1_stride lo
    g1[7] = 0;
    i32x4 z4 = {};
#if __clang_major__ >= 23
    i32x8 z8 = {};
    __builtin_amdgcn_tensor_load_to_lds(g0, g1, z4, z4, z8, 0);
#else
    __builtin_amdgcn_tensor_load_to_lds(g0, g1, z4, z4, 0);
#endif
}
#endif

// ---------------------------------------------------------------------------
// Precision-conversion kernels (run once; everything downstream is bf16).
// ---------------------------------------------------------------------------
__global__ __launch_bounds__(256) void cvt_bf16_kernel(
    const float* __restrict__ s, __bf16* __restrict__ d, int n4) {
    int i = blockIdx.x * 256 + threadIdx.x;
    if (i >= n4) return;
    v4f v = ((const v4f*)s)[i];
    v4bf o;
    o[0] = (__bf16)v[0]; o[1] = (__bf16)v[1];
    o[2] = (__bf16)v[2]; o[3] = (__bf16)v[3];
    ((v4bf*)d)[i] = o;
}

// D[col][row] = (bf16) S[row][col]   (coalesced both sides via LDS tile)
__global__ __launch_bounds__(256) void transpose_cvt_kernel(
    const float* __restrict__ S, __bf16* __restrict__ D, int rows, int cols) {
    __shared__ float tile[32][33];
    int bx = blockIdx.x * 32, by = blockIdx.y * 32;
    int tx = threadIdx.x & 31, ty = threadIdx.x >> 5;   // ty in 0..7
    #pragma unroll
    for (int i = 0; i < 32; i += 8)
        tile[ty + i][tx] = S[(size_t)(by + ty + i) * cols + bx + tx];
    __syncthreads();
    #pragma unroll
    for (int i = 0; i < 32; i += 8)
        D[(size_t)(bx + ty + i) * rows + by + tx] = (__bf16)tile[tx][ty + i];
}

// ---------------------------------------------------------------------------
// Kernel 1: qkv = Xb @ Wqt^T + b ; scatter Q,K row-major [B,H,T,64], V^T [B,H,64,T].
// Block tile 128(M) x 64(N), K-step 64, 8 waves (4x2), each wave 32x32.
// A tile via TDM (fallback: async b128), B tile via async b128.
// ---------------------------------------------------------------------------
__global__ __launch_bounds__(256) void qkv_gemm_kernel(
    const __bf16* __restrict__ Xb, const __bf16* __restrict__ Wt,
    const float* __restrict__ bias,
    __bf16* __restrict__ Qb, __bf16* __restrict__ Kb, __bf16* __restrict__ Vt)
{
    __shared__ __bf16 a_lds[128 * 72];   // rows padded to 72 (144B)
    __shared__ __bf16 b_lds[64 * 72];    // n-major

    const int tid  = threadIdx.x;
    const int lane = tid & 31;
    const int w    = tid >> 5;
    const int wm   = w & 3;
    const int wn   = w >> 2;
    const int mBase = blockIdx.y * 128;
    const int nBase = blockIdx.x * 64;
    const unsigned a_base = (unsigned)(uintptr_t)(void*)a_lds;
    const unsigned b_base = (unsigned)(uintptr_t)(void*)b_lds;

    v8f acc[2][2] = {};

    for (int k0 = 0; k0 < NEMBD; k0 += 64) {
#if HAS_TDM
        if (tid == 0)
            tdm_load_tile_a(Xb + (size_t)mBase * NEMBD + k0, a_base);
#else
        stage_rows_async(Xb + (size_t)mBase * NEMBD + k0, NEMBD, a_base, 1024, tid);
#endif
        stage_rows_async(Wt + (size_t)nBase * NEMBD + k0, NEMBD, b_base, 512, tid);
        wait_async0();
#if HAS_TDM
        if (tid == 0) wait_tensor0();
#endif
        __syncthreads();

        #pragma unroll
        for (int kc = 0; kc < 2; ++kc) {
            #pragma unroll
            for (int mi = 0; mi < 2; ++mi) {
                v16bf af = load_frag_a(a_lds + (wm * 32 + mi * 16) * 72 + kc * 32, 72, lane);
                #pragma unroll
                for (int ni = 0; ni < 2; ++ni) {
                    v16bf bf = load_frag_b(b_lds + (wn * 32 + ni * 16) * 72 + kc * 32, 72, lane);
                    acc[mi][ni] = wmma_bf16(af, bf, acc[mi][ni]);
                }
            }
        }
        __syncthreads();
    }

    // epilogue: bias + scatter to Q / K / V^T
    #pragma unroll
    for (int mi = 0; mi < 2; ++mi) {
        #pragma unroll
        for (int ni = 0; ni < 2; ++ni) {
            int col = nBase + wn * 32 + ni * 16 + (lane & 15);
            float bv = bias[col];
            int which = col >> 10;          // 0=q 1=k 2=v
            int c = col & 1023;
            int h = c >> 6, d = c & 63;
            #pragma unroll
            for (int r = 0; r < 8; ++r) {
                int row = mBase + wm * 32 + mi * 16 + r + ((lane & 16) ? 8 : 0);
                int bb = row >> 11, t = row & (TSEQ - 1);
                float val = acc[mi][ni][r] + bv;
                size_t head = (size_t)bb * NHEAD + h;
                if (which == 0)
                    Qb[(head * TSEQ + t) * HDIM + d] = (__bf16)val;
                else if (which == 1)
                    Kb[(head * TSEQ + t) * HDIM + d] = (__bf16)val;
                else
                    Vt[(head * HDIM + d) * TSEQ + t] = (__bf16)val;   // transposed
            }
        }
    }
}

// ---------------------------------------------------------------------------
// Kernel 2: causal flash attention.  One wave = 16 query rows; block = 8 waves.
// grid = (B*H, T/128).  Q,K row-major [T,64] bf16; V transposed [64,T] bf16.
// ---------------------------------------------------------------------------
__global__ __launch_bounds__(256) void flash_attn_kernel(
    const __bf16* __restrict__ Qb, const __bf16* __restrict__ Kb,
    const __bf16* __restrict__ Vt, __bf16* __restrict__ Y)
{
    __shared__ __bf16 p_lds[8][16 * 40];     // per-wave P patch (16x32, padded)

    const int lane = threadIdx.x & 31;
    const int w    = threadIdx.x >> 5;
    const int bh   = blockIdx.x;
    const int qr   = blockIdx.y * 128 + w * 16;

    const __bf16* Qh = Qb + (size_t)bh * TSEQ * HDIM;
    const __bf16* Kh = Kb + (size_t)bh * TSEQ * HDIM;
    const __bf16* Vh = Vt + (size_t)bh * HDIM * TSEQ;

    v16bf qf[2];
    #pragma unroll
    for (int c = 0; c < 2; ++c)
        qf[c] = load_frag_a(Qh + (size_t)qr * HDIM + c * 32, HDIM, lane);

    v8f o[4] = {};
    float m_r[8], l_r[8];
    #pragma unroll
    for (int r = 0; r < 8; ++r) { m_r[r] = -INFINITY; l_r[r] = 0.0f; }

    const float scale = 0.125f;   // 1/sqrt(64)
    const int rowoff = (lane & 16) ? 8 : 0;
    const int nkb = ((qr + 15) >> 5) + 1;

    for (int kb = 0; kb < nkb; ++kb) {
        const int kbase = kb * 32;

        v8f s[2];
        #pragma unroll
        for (int nt = 0; nt < 2; ++nt) {
            v8f a = {};
            #pragma unroll
            for (int c = 0; c < 2; ++c) {
                v16bf kf = load_frag_b(Kh + (size_t)(kbase + nt * 16) * HDIM + c * 32,
                                       HDIM, lane);
                a = wmma_bf16(qf[c], kf, a);
            }
            s[nt] = a;
        }

        float alpha[8];
        #pragma unroll
        for (int r = 0; r < 8; ++r) {
            int row = qr + r + rowoff;
            int key0 = kbase + (lane & 15);
            int key1 = key0 + 16;
            float f0 = (key0 <= row) ? s[0][r] * scale : -INFINITY;
            float f1 = (key1 <= row) ? s[1][r] * scale : -INFINITY;
            float mx = fmaxf(f0, f1);
            #pragma unroll
            for (int off = 1; off < 16; off <<= 1)
                mx = fmaxf(mx, __shfl_xor(mx, off, 16));
            float mn = fmaxf(m_r[r], mx);
            alpha[r] = __expf(m_r[r] - mn);
            m_r[r] = mn;
            float p0 = __expf(f0 - mn);
            float p1 = __expf(f1 - mn);
            s[0][r] = p0; s[1][r] = p1;
            float sum = p0 + p1;
            #pragma unroll
            for (int off = 1; off < 16; off <<= 1)
                sum += __shfl_xor(sum, off, 16);
            l_r[r] = l_r[r] * alpha[r] + sum;
        }

        __bf16* pl = p_lds[w];
        #pragma unroll
        for (int nt = 0; nt < 2; ++nt) {
            #pragma unroll
            for (int r = 0; r < 8; ++r)
                pl[(r + rowoff) * 40 + nt * 16 + (lane & 15)] = (__bf16)s[nt][r];
        }
        v16bf pf = load_frag_a(pl, 40, lane);

        #pragma unroll
        for (int nd = 0; nd < 4; ++nd) {
            #pragma unroll
            for (int r = 0; r < 8; ++r) o[nd][r] *= alpha[r];
            v16bf vf = load_frag_b(Vh + (size_t)(nd * 16) * TSEQ + kbase, TSEQ, lane);
            o[nd] = wmma_bf16(pf, vf, o[nd]);
        }
    }

    const int h = bh & (NHEAD - 1);
    const int b = bh >> 4;
    #pragma unroll
    for (int nd = 0; nd < 4; ++nd) {
        int col = h * HDIM + nd * 16 + (lane & 15);
        #pragma unroll
        for (int r = 0; r < 8; ++r) {
            int row = qr + r + rowoff;
            size_t tg = (size_t)b * TSEQ + row;
            Y[tg * NEMBD + col] = (__bf16)(o[nd][r] / l_r[r]);
        }
    }
}

// ---------------------------------------------------------------------------
// Kernel 3: out = Yb @ Wot^T + b_o  (out f32)
// ---------------------------------------------------------------------------
__global__ __launch_bounds__(256) void oproj_gemm_kernel(
    const __bf16* __restrict__ Yb, const __bf16* __restrict__ Wt,
    const float* __restrict__ bias, float* __restrict__ out)
{
    __shared__ __bf16 a_lds[128 * 72];
    __shared__ __bf16 b_lds[64 * 72];

    const int tid  = threadIdx.x;
    const int lane = tid & 31;
    const int w    = tid >> 5;
    const int wm   = w & 3;
    const int wn   = w >> 2;
    const int mBase = blockIdx.y * 128;
    const int nBase = blockIdx.x * 64;
    const unsigned a_base = (unsigned)(uintptr_t)(void*)a_lds;
    const unsigned b_base = (unsigned)(uintptr_t)(void*)b_lds;

    v8f acc[2][2] = {};

    for (int k0 = 0; k0 < NEMBD; k0 += 64) {
#if HAS_TDM
        if (tid == 0)
            tdm_load_tile_a(Yb + (size_t)mBase * NEMBD + k0, a_base);
#else
        stage_rows_async(Yb + (size_t)mBase * NEMBD + k0, NEMBD, a_base, 1024, tid);
#endif
        stage_rows_async(Wt + (size_t)nBase * NEMBD + k0, NEMBD, b_base, 512, tid);
        wait_async0();
#if HAS_TDM
        if (tid == 0) wait_tensor0();
#endif
        __syncthreads();

        #pragma unroll
        for (int kc = 0; kc < 2; ++kc) {
            #pragma unroll
            for (int mi = 0; mi < 2; ++mi) {
                v16bf af = load_frag_a(a_lds + (wm * 32 + mi * 16) * 72 + kc * 32, 72, lane);
                #pragma unroll
                for (int ni = 0; ni < 2; ++ni) {
                    v16bf bf = load_frag_b(b_lds + (wn * 32 + ni * 16) * 72 + kc * 32, 72, lane);
                    acc[mi][ni] = wmma_bf16(af, bf, acc[mi][ni]);
                }
            }
        }
        __syncthreads();
    }

    #pragma unroll
    for (int mi = 0; mi < 2; ++mi) {
        #pragma unroll
        for (int ni = 0; ni < 2; ++ni) {
            int col = nBase + wn * 32 + ni * 16 + (lane & 15);
            float bv = bias[col];
            #pragma unroll
            for (int r = 0; r < 8; ++r) {
                int row = mBase + wm * 32 + mi * 16 + r + ((lane & 16) ? 8 : 0);
                out[(size_t)row * NEMBD + col] = acc[mi][ni][r] + bv;
            }
        }
    }
}

// ---------------------------------------------------------------------------
extern "C" void kernel_launch(void* const* d_in, const int* in_sizes, int n_in,
                              void* d_out, int out_size, void* d_ws, size_t ws_size,
                              hipStream_t stream) {
    const float* x     = (const float*)d_in[0];
    const float* W_qkv = (const float*)d_in[1];
    const float* b_qkv = (const float*)d_in[2];
    const float* W_o   = (const float*)d_in[3];
    const float* b_o   = (const float*)d_in[4];
    float* out = (float*)d_out;

    char* ws = (char*)d_ws;
    const size_t SZ = (size_t)BT * NEMBD * sizeof(__bf16);   // 8 MiB
    __bf16* Qb  = (__bf16*)(ws);
    __bf16* Kb  = (__bf16*)(ws + SZ);
    __bf16* Vt  = (__bf16*)(ws + 2 * SZ);
    __bf16* Yb  = (__bf16*)(ws + 3 * SZ);
    __bf16* Xb  = (__bf16*)(ws + 4 * SZ);                    // [4096][1024]
    __bf16* Wqt = (__bf16*)(ws + 5 * SZ);                    // [3072][1024] (6 MiB)
    __bf16* Wot = (__bf16*)(ws + 5 * SZ + (size_t)3 * NEMBD * NEMBD * sizeof(__bf16));

    // one-time bf16 conversion (+ weight transposes to n-major)
    cvt_bf16_kernel<<<(BT * NEMBD / 4 + 255) / 256, 256, 0, stream>>>(x, Xb, BT * NEMBD / 4);
    transpose_cvt_kernel<<<dim3(96, 32), 256, 0, stream>>>(W_qkv, Wqt, NEMBD, 3 * NEMBD);
    transpose_cvt_kernel<<<dim3(32, 32), 256, 0, stream>>>(W_o, Wot, NEMBD, NEMBD);

    // qkv: M=4096, N=3072
    qkv_gemm_kernel<<<dim3(48, 32), 256, 0, stream>>>(Xb, Wqt, b_qkv, Qb, Kb, Vt);
    // attention: grid (B*H, T/128)
    flash_attn_kernel<<<dim3(32, 16), 256, 0, stream>>>(Qb, Kb, Vt, Yb);
    // o-proj: M=4096, N=1024
    oproj_gemm_kernel<<<dim3(16, 32), 256, 0, stream>>>(Yb, Wot, b_o, out);
}
